// VQVAE_11106785427823
// MI455X (gfx1250) — matrix operand order repaired and checked by reading, not compiled, not run on hardware
//
#include <hip/hip_runtime.h>
#include <hip/hip_bf16.h>

// ---------------------------------------------------------------------------
// VQ-VAE forward for MI455X (gfx1250, wave32, WMMA).
// All GEMMs use v_wmma_f32_16x16x32_f16 (f16 inputs, f32 accumulate).
// ---------------------------------------------------------------------------

typedef __attribute__((ext_vector_type(16))) _Float16 v16h;
typedef __attribute__((ext_vector_type(8)))  _Float16 v8h;
typedef __attribute__((ext_vector_type(8)))  float    v8f;

#define B_N   32768
#define D_IN  512
#define H1_N  256
#define H2_N  128
#define LAT_N 64
#define K_CB  2048
#define BN_EPS 1e-5f

// ---------------------------- prep kernels ---------------------------------

__global__ __launch_bounds__(256) void k_cvt_f16(const float* __restrict__ src,
                                                 _Float16* __restrict__ dst,
                                                 long n) {
  long stride = (long)gridDim.x * blockDim.x;
  for (long i = (long)blockIdx.x * blockDim.x + threadIdx.x; i < n; i += stride)
    dst[i] = (_Float16)src[i];
}

// W is [K][N] f32 row-major; Wt is [N][K] f16 row-major (transposed).
__global__ __launch_bounds__(256) void k_transpose_f16(const float* __restrict__ W,
                                                       _Float16* __restrict__ Wt,
                                                       int K, int N) {
  long t = (long)blockIdx.x * blockDim.x + threadIdx.x;
  if (t >= (long)K * N) return;
  int n = (int)(t / K);
  int k = (int)(t % K);
  Wt[t] = (_Float16)W[(long)k * N + n];
}

// fold BN (eval) into scale/shift: y = x*s + t, s = g*rsqrt(v+eps), t = b - m*s
__global__ void k_bnprep(const float* __restrict__ g, const float* __restrict__ b,
                         const float* __restrict__ m, const float* __restrict__ v,
                         float* __restrict__ s, float* __restrict__ t, int n) {
  int i = blockIdx.x * blockDim.x + threadIdx.x;
  if (i >= n) return;
  float sc = g[i] * rsqrtf(v[i] + BN_EPS);
  s[i] = sc;
  t[i] = b[i] - m[i] * sc;
}

// per-code squared norms, codebook is [K_CB][LAT] f32
__global__ void k_cn2(const float* __restrict__ cb, float* __restrict__ cn2) {
  int k = blockIdx.x * blockDim.x + threadIdx.x;
  if (k >= K_CB) return;
  const float* row = cb + (long)k * LAT_N;
  float acc = 0.f;
  for (int d = 0; d < LAT_N; ++d) acc += row[d] * row[d];
  cn2[k] = acc;
}

__global__ void k_zero2(float* __restrict__ acc) {
  if (threadIdx.x == 0) { acc[0] = 0.f; acc[1] = 0.f; }
}

// ------------------------------ GEMM ---------------------------------------
// One wave computes one 16x16 output tile of  out[M,N] = A[M,K] @ Wt[N,K]^T.
// A operand layout (ISA 7.12.2, 16-bit A 16x32): lane L holds row (L&15),
//   halves 0..7 = K (L>=16 ? 8..15 : 0..7), halves 8..15 = +16.
// B operand: lane L holds column (L&15), same K order as A.
// C/D: VGPR i, lanes 0-15 -> (M = m0+i, N = lane), lanes 16-31 -> M = m0+8+i.
//
// mode 0: y = relu(acc+bias)*s + t            -> out16
// mode 1: y = acc+bias                        -> out16 and out32   (z_e)
// mode 2: y = acc+bias                        -> out32; accumulate sum((y-xref)^2)

__global__ __launch_bounds__(256)
void k_gemm(const _Float16* __restrict__ A, const _Float16* __restrict__ Wt,
            const float* __restrict__ bias, const float* __restrict__ s,
            const float* __restrict__ t, _Float16* __restrict__ out16,
            float* __restrict__ out32, const float* __restrict__ xref,
            float* __restrict__ accum, int M, int Kd, int N, int mode) {
  const int wave = threadIdx.x >> 5;
  const int ln   = threadIdx.x & 31;
  const int ntiles = N >> 4;
  const int tile = blockIdx.x * 8 + wave;
  if (tile >= (M >> 4) * ntiles) return;           // wave-uniform guard
  const int m0 = (tile / ntiles) << 4;
  const int n0 = (tile % ntiles) << 4;

  const int rowA  = m0 + (ln & 15);
  const int koffA = (ln >> 4) * 8;
  const int colB  = n0 + (ln & 15);
  const int koffB = (ln >> 4) * 8;
  const _Float16* __restrict__ Ap = A  + (long)rowA * Kd;
  const _Float16* __restrict__ Bp = Wt + (long)colB * Kd;

  v8f acc = {};
  for (int k0 = 0; k0 < Kd; k0 += 32) {
    v8h al = *(const v8h*)(Ap + k0 + koffA);
    v8h ah = *(const v8h*)(Ap + k0 + koffA + 16);
    v8h bl = *(const v8h*)(Bp + k0 + koffB);
    v8h bh = *(const v8h*)(Bp + k0 + koffB + 16);
    v16h a, b;
#pragma unroll
    for (int i = 0; i < 8; ++i) {
      a[i] = al[i]; a[i + 8] = ah[i];
      b[i] = bl[i]; b[i + 8] = bh[i];
    }
    acc = __builtin_amdgcn_wmma_f32_16x16x32_f16(false, a, false, b,
                                                 (short)0, acc, false, false);
  }

  const int col   = n0 + (ln & 15);
  const int rbase = m0 + ((ln >> 4) << 3);
  const float bb = bias[col];

  if (mode == 0) {
    const float sc = s[col], tb = t[col];
#pragma unroll
    for (int i = 0; i < 8; ++i) {
      float y = acc[i] + bb;
      y = y > 0.f ? y : 0.f;
      y = y * sc + tb;
      out16[(long)(rbase + i) * N + col] = (_Float16)y;
    }
  } else if (mode == 1) {
#pragma unroll
    for (int i = 0; i < 8; ++i) {
      float y = acc[i] + bb;
      long o = (long)(rbase + i) * N + col;
      out16[o] = (_Float16)y;
      out32[o] = y;
    }
  } else {
    float part = 0.f;
#pragma unroll
    for (int i = 0; i < 8; ++i) {
      float y = acc[i] + bb;
      long o = (long)(rbase + i) * N + col;
      out32[o] = y;
      float d = y - xref[o];
      part += d * d;
    }
#pragma unroll
    for (int off = 16; off >= 1; off >>= 1)
      part += __shfl_xor(part, off, 32);
    if (ln == 0) atomicAdd(accum + 1, part);
  }
}

// ------------------------------ VQ argmin ----------------------------------
// One wave handles 16 rows of z_e; sweeps all K_CB/16 code tiles with WMMA,
// tracking running min of  ||c||^2 - 2 * z_e . c  (argmin-equivalent to d2).

__global__ __launch_bounds__(32)
void k_vq(const _Float16* __restrict__ ze16, const _Float16* __restrict__ cb16,
          const float* __restrict__ cn2, int* __restrict__ idx_i,
          float* __restrict__ idx_f) {
  const int ln = threadIdx.x & 31;
  const int m0 = blockIdx.x << 4;
  const int rowA  = m0 + (ln & 15);
  const int koffA = (ln >> 4) * 8;
  const int koffB = (ln >> 4) * 8;
  const _Float16* __restrict__ zp = ze16 + (long)rowA * LAT_N;

  v16h a0, a1;
  {
    v8h l0 = *(const v8h*)(zp + koffA);
    v8h h0 = *(const v8h*)(zp + koffA + 16);
    v8h l1 = *(const v8h*)(zp + 32 + koffA);
    v8h h1 = *(const v8h*)(zp + 32 + koffA + 16);
#pragma unroll
    for (int i = 0; i < 8; ++i) {
      a0[i] = l0[i]; a0[i + 8] = h0[i];
      a1[i] = l1[i]; a1[i + 8] = h1[i];
    }
  }

  float best[8];
  int bidx[8];
#pragma unroll
  for (int i = 0; i < 8; ++i) { best[i] = 3.4e38f; bidx[i] = 0; }

  for (int nt = 0; nt < (K_CB >> 4); ++nt) {
    const int n = (nt << 4) + (ln & 15);
    const _Float16* __restrict__ cp = cb16 + (long)n * LAT_N;
    v16h b0, b1;
    v8h l0 = *(const v8h*)(cp + koffB);
    v8h h0 = *(const v8h*)(cp + koffB + 16);
    v8h l1 = *(const v8h*)(cp + 32 + koffB);
    v8h h1 = *(const v8h*)(cp + 32 + koffB + 16);
#pragma unroll
    for (int i = 0; i < 8; ++i) {
      b0[i] = l0[i]; b0[i + 8] = h0[i];
      b1[i] = l1[i]; b1[i + 8] = h1[i];
    }
    v8f acc = {};
    acc = __builtin_amdgcn_wmma_f32_16x16x32_f16(false, a0, false, b0,
                                                 (short)0, acc, false, false);
    acc = __builtin_amdgcn_wmma_f32_16x16x32_f16(false, a1, false, b1,
                                                 (short)0, acc, false, false);
    const float c2 = cn2[n];
#pragma unroll
    for (int i = 0; i < 8; ++i) {
      float d2 = c2 - 2.f * acc[i];
      if (d2 < best[i]) { best[i] = d2; bidx[i] = n; }
    }
  }

  // reduce across the 16 lanes of each half (xor masks stay inside halves)
#pragma unroll
  for (int off = 1; off < 16; off <<= 1) {
#pragma unroll
    for (int i = 0; i < 8; ++i) {
      float ov = __shfl_xor(best[i], off, 32);
      int   oi = __shfl_xor(bidx[i], off, 32);
      if (ov < best[i] || (ov == best[i] && oi < bidx[i])) {
        best[i] = ov; bidx[i] = oi;
      }
    }
  }
  if (ln == 0) {
#pragma unroll
    for (int i = 0; i < 8; ++i) {
      idx_i[m0 + i] = bidx[i];
      idx_f[m0 + i] = (float)bidx[i];
    }
  }
  if (ln == 16) {
#pragma unroll
    for (int i = 0; i < 8; ++i) {
      idx_i[m0 + 8 + i] = bidx[i];
      idx_f[m0 + 8 + i] = (float)bidx[i];
    }
  }
}

// ------------------- gather z_q + vq loss accumulation ---------------------

__global__ __launch_bounds__(256)
void k_gather(const int* __restrict__ idx, const float* __restrict__ cb,
              const float* __restrict__ ze32, _Float16* __restrict__ zq16,
              float* __restrict__ accum) {
  long t = (long)blockIdx.x * blockDim.x + threadIdx.x;   // t < B*LAT
  int b = (int)(t >> 6), d = (int)(t & 63);
  float c = cb[(long)idx[b] * LAT_N + d];
  zq16[t] = (_Float16)c;
  float diff = c - ze32[t];
  float p = diff * diff;
#pragma unroll
  for (int off = 16; off >= 1; off >>= 1) p += __shfl_xor(p, off, 32);
  if ((threadIdx.x & 31) == 0) atomicAdd(accum, p);
}

__global__ void k_final(const float* __restrict__ accum,
                        float* __restrict__ vq_out, float* __restrict__ rec_out) {
  if (threadIdx.x == 0) {
    *vq_out  = 1.25f * accum[0] / (float)((long)B_N * LAT_N);
    *rec_out = accum[1] / (float)((long)B_N * D_IN);
  }
}

// ------------------------------ launch -------------------------------------

extern "C" void kernel_launch(void* const* d_in, const int* in_sizes, int n_in,
                              void* d_out, int out_size, void* d_ws, size_t ws_size,
                              hipStream_t stream) {
  (void)in_sizes; (void)n_in; (void)out_size; (void)ws_size;
  const float* x    = (const float*)d_in[0];
  const float* ew1  = (const float*)d_in[1];  const float* eb1 = (const float*)d_in[2];
  const float* eg1  = (const float*)d_in[3];  const float* ebt1= (const float*)d_in[4];
  const float* em1  = (const float*)d_in[5];  const float* ev1 = (const float*)d_in[6];
  const float* ew2  = (const float*)d_in[7];  const float* eb2 = (const float*)d_in[8];
  const float* eg2  = (const float*)d_in[9];  const float* ebt2= (const float*)d_in[10];
  const float* em2  = (const float*)d_in[11]; const float* ev2 = (const float*)d_in[12];
  const float* ew3  = (const float*)d_in[13]; const float* eb3 = (const float*)d_in[14];
  const float* cb   = (const float*)d_in[15];
  const float* dw1  = (const float*)d_in[16]; const float* db1 = (const float*)d_in[17];
  const float* dg1  = (const float*)d_in[18]; const float* dbt1= (const float*)d_in[19];
  const float* dm1  = (const float*)d_in[20]; const float* dv1 = (const float*)d_in[21];
  const float* dw2  = (const float*)d_in[22]; const float* db2 = (const float*)d_in[23];
  const float* dg2  = (const float*)d_in[24]; const float* dbt2= (const float*)d_in[25];
  const float* dm2  = (const float*)d_in[26]; const float* dv2 = (const float*)d_in[27];
  const float* dw3  = (const float*)d_in[28]; const float* db3 = (const float*)d_in[29];

  float* out = (float*)d_out;
  float* xrecon   = out;                               // [B, 512]
  float* vq_loss  = out + (long)B_N * D_IN;            // [1]
  float* idx_f    = vq_loss + 1;                       // [B]
  float* recon_err= idx_f + B_N;                       // [1]

  // workspace carve-up
  char* ws = (char*)d_ws;
  size_t off = 0;
  auto carve = [&](size_t bytes) -> char* {
    char* p = ws + off;
    off = (off + bytes + 255) & ~(size_t)255;
    return p;
  };
  _Float16* ew1T = (_Float16*)carve((size_t)H1_N * D_IN * 2);
  _Float16* ew2T = (_Float16*)carve((size_t)H2_N * H1_N * 2);
  _Float16* ew3T = (_Float16*)carve((size_t)LAT_N * H2_N * 2);
  _Float16* dw1T = (_Float16*)carve((size_t)H2_N * LAT_N * 2);
  _Float16* dw2T = (_Float16*)carve((size_t)H1_N * H2_N * 2);
  _Float16* dw3T = (_Float16*)carve((size_t)D_IN * H1_N * 2);
  _Float16* cb16 = (_Float16*)carve((size_t)K_CB * LAT_N * 2);
  float* cn2 = (float*)carve((size_t)K_CB * 4);
  float* s1  = (float*)carve(H1_N * 4);  float* t1  = (float*)carve(H1_N * 4);
  float* s2  = (float*)carve(H2_N * 4);  float* t2  = (float*)carve(H2_N * 4);
  float* sd1 = (float*)carve(H2_N * 4);  float* td1 = (float*)carve(H2_N * 4);
  float* sd2 = (float*)carve(H1_N * 4);  float* td2 = (float*)carve(H1_N * 4);
  _Float16* x16  = (_Float16*)carve((size_t)B_N * D_IN * 2);
  _Float16* bufA = (_Float16*)carve((size_t)B_N * H1_N * 2);  // h1, then dh2
  _Float16* bufB = (_Float16*)carve((size_t)B_N * H2_N * 2);  // h2, then dh1
  _Float16* ze16 = (_Float16*)carve((size_t)B_N * LAT_N * 2);
  _Float16* zq16 = (_Float16*)carve((size_t)B_N * LAT_N * 2);
  float*    ze32 = (float*)carve((size_t)B_N * LAT_N * 4);
  int*      idxi = (int*)carve((size_t)B_N * 4);
  float*    accum= (float*)carve(8);                 // [0]=vq sum, [1]=recon sum

  // ---- prep ----
  k_zero2<<<1, 32, 0, stream>>>(accum);
  k_cvt_f16<<<4096, 256, 0, stream>>>(x,  x16,  (long)B_N * D_IN);
  k_cvt_f16<<<512,  256, 0, stream>>>(cb, cb16, (long)K_CB * LAT_N);
  k_transpose_f16<<<(D_IN*H1_N)/256,  256, 0, stream>>>(ew1, ew1T, D_IN,  H1_N);
  k_transpose_f16<<<(H1_N*H2_N)/256,  256, 0, stream>>>(ew2, ew2T, H1_N,  H2_N);
  k_transpose_f16<<<(H2_N*LAT_N)/256, 256, 0, stream>>>(ew3, ew3T, H2_N,  LAT_N);
  k_transpose_f16<<<(LAT_N*H2_N)/256, 256, 0, stream>>>(dw1, dw1T, LAT_N, H2_N);
  k_transpose_f16<<<(H2_N*H1_N)/256,  256, 0, stream>>>(dw2, dw2T, H2_N,  H1_N);
  k_transpose_f16<<<(H1_N*D_IN)/256,  256, 0, stream>>>(dw3, dw3T, H1_N,  D_IN);
  k_bnprep<<<1, H1_N, 0, stream>>>(eg1, ebt1, em1, ev1, s1, t1, H1_N);
  k_bnprep<<<1, H2_N, 0, stream>>>(eg2, ebt2, em2, ev2, s2, t2, H2_N);
  k_bnprep<<<1, H2_N, 0, stream>>>(dg1, dbt1, dm1, dv1, sd1, td1, H2_N);
  k_bnprep<<<1, H1_N, 0, stream>>>(dg2, dbt2, dm2, dv2, sd2, td2, H1_N);
  k_cn2<<<K_CB/256, 256, 0, stream>>>(cb, cn2);

  // ---- encoder ----
  k_gemm<<<(B_N/16)*(H1_N/16)/8, 256, 0, stream>>>(x16,  ew1T, eb1, s1, t1,
         bufA, nullptr, nullptr, nullptr, B_N, D_IN, H1_N, 0);
  k_gemm<<<(B_N/16)*(H2_N/16)/8, 256, 0, stream>>>(bufA, ew2T, eb2, s2, t2,
         bufB, nullptr, nullptr, nullptr, B_N, H1_N, H2_N, 0);
  k_gemm<<<(B_N/16)*(LAT_N/16)/8, 256, 0, stream>>>(bufB, ew3T, eb3, nullptr, nullptr,
         ze16, ze32, nullptr, nullptr, B_N, H2_N, LAT_N, 1);

  // ---- vector quantization ----
  k_vq<<<B_N/16, 32, 0, stream>>>(ze16, cb16, cn2, idxi, idx_f);
  k_gather<<<(B_N*LAT_N)/256, 256, 0, stream>>>(idxi, cb, ze32, zq16, accum);

  // ---- decoder ----
  k_gemm<<<(B_N/16)*(H2_N/16)/8, 256, 0, stream>>>(zq16, dw1T, db1, sd1, td1,
         bufB, nullptr, nullptr, nullptr, B_N, LAT_N, H2_N, 0);
  k_gemm<<<(B_N/16)*(H1_N/16)/8, 256, 0, stream>>>(bufB, dw2T, db2, sd2, td2,
         bufA, nullptr, nullptr, nullptr, B_N, H2_N, H1_N, 0);
  k_gemm<<<(B_N/16)*(D_IN/16)/8, 256, 0, stream>>>(bufA, dw3T, db3, nullptr, nullptr,
         nullptr, xrecon, x, accum, B_N, H1_N, D_IN, 2);

  // ---- scalar outputs ----
  k_final<<<1, 32, 0, stream>>>(accum, vq_loss, recon_err);
}
